// SlotAttention_42236708389328
// MI455X (gfx1250) — compile-verified, matrix operand order checked by reading, not run on hardware
//
#include <hip/hip_runtime.h>
#include <hip/hip_bf16.h>
#include <math.h>

#define B_ 16
#define T_ 4
#define N_ 4096
#define D_ 128
#define S_ 8
#define H_ 128
#define BT_ (B_*T_)
#define ITERS_ 3
#define LN_EPS_ 1e-5f
#define EPS_ 1e-8f
#define SCALE_ 0.08838834764831845f   // 128^-0.5
#define PITCH_ 136                    // 128 + 8 halfs pad (272B rows, conflict-free)
#define NCH_ 8                        // split-K chunks for attn@v
#define JCH_ (N_/NCH_)                // 512 j's per chunk

typedef __attribute__((ext_vector_type(16))) _Float16 v16h;
typedef __attribute__((ext_vector_type(8)))  _Float16 v8h;
typedef __attribute__((ext_vector_type(8)))  float    v8f;

// ---- WMMA fragment helpers (16-bit A/B layout, ISA 7.12.2) -----------------
// A 16x32 f16: lane L holds row M=L%16; chunk0 K=[half*8, +8), chunk1 K=[16+half*8, +8)
// B 32x16 f16 (mirrored): lane L holds column N=L%16 with the same K chunking,
// so a column of B == a contiguous row of the (pre-transposed) source matrix.
__device__ __forceinline__ v16h frag_ld(const _Float16* rowbase, int k0, int half) {
  const _Float16* p = rowbase + k0 + half * 8;
  v8h lo = *(const v8h*)(p);
  v8h hi = *(const v8h*)(p + 16);
  v16h f;
#pragma unroll
  for (int i = 0; i < 8; ++i) { f[i] = lo[i]; f[i + 8] = hi[i]; }
  return f;
}

__device__ __forceinline__ v8f wmma16(v16h a, v16h b, v8f c) {
  return __builtin_amdgcn_wmma_f32_16x16x32_f16(false, a, false, b, (short)0, c, false, false);
}

__device__ __forceinline__ v16h frag_zero() {
  v16h z;
#pragma unroll
  for (int i = 0; i < 16; ++i) z[i] = (_Float16)0.f;
  return z;
}

__device__ __forceinline__ float sigmoidf_(float x) { return 1.f / (1.f + __expf(-x)); }

// ---- K0: f32 -> f16 weight conversion --------------------------------------
__global__ void k_cvt_f16(const float* __restrict__ s, _Float16* __restrict__ d, int n) {
  int i = blockIdx.x * 256 + threadIdx.x;
  if (i < n) d[i] = (_Float16)s[i];
}

// ---- K1: slots = mu + exp(logsigma) * noise --------------------------------
__global__ void k_init_slots(const float* __restrict__ noise, const float* __restrict__ mu,
                             const float* __restrict__ lsig, float* __restrict__ slots) {
  int idx = blockIdx.x * 256 + threadIdx.x;           // over BT_*S_*D_ = 65536
  int dd = idx & (D_ - 1);
  int bt = idx / (S_ * D_);
  int t  = bt & (T_ - 1);
  slots[idx] = mu[t * D_ + dd] + __expf(lsig[t * D_ + dd]) * noise[idx];
}

// ---- K2: LN(inputs) then k = x@Wk^T+bk (row-major f16), v = x@Wv^T+bv (transposed f16)
__global__ void k_ln_kv(const float* __restrict__ x, const float* __restrict__ g,
                        const float* __restrict__ beta,
                        const _Float16* __restrict__ WkH, const _Float16* __restrict__ WvH,
                        const float* __restrict__ bk, const float* __restrict__ bv,
                        _Float16* __restrict__ kh, _Float16* __restrict__ vt) {
  __shared__ __align__(16) _Float16 xh[16 * PITCH_];
  const int tid = threadIdx.x, lane = tid & 31, wave = tid >> 5;
  const int l16 = lane & 15, half = lane >> 4;
  const int row0 = blockIdx.x * 16;                    // global row over (b,t,n)

  // LN: wave w handles rows 2w+half; 16 lanes x 8 elems per row
  {
    int row = 2 * wave + half;
    const float* src = x + (size_t)(row0 + row) * D_ + l16 * 8;
    float v[8], s = 0.f, sq = 0.f;
#pragma unroll
    for (int c = 0; c < 8; ++c) { v[c] = src[c]; s += v[c]; sq += v[c] * v[c]; }
#pragma unroll
    for (int m = 1; m < 16; m <<= 1) { s += __shfl_xor(s, m, 32); sq += __shfl_xor(sq, m, 32); }
    float mean = s * (1.f / D_);
    float var  = sq * (1.f / D_) - mean * mean;
    float inv  = rsqrtf(var + LN_EPS_);
#pragma unroll
    for (int c = 0; c < 8; ++c) {
      int col = l16 * 8 + c;
      xh[row * PITCH_ + col] = (_Float16)((v[c] - mean) * inv * g[col] + beta[col]);
    }
  }
  __syncthreads();

  // wave w -> output column tile w (16 cols) for both k and v
  const int colb = wave * 16;
  const _Float16* Arow = xh + l16 * PITCH_;
  const _Float16* Bk = WkH + (size_t)(colb + l16) * D_;
  const _Float16* Bv = WvH + (size_t)(colb + l16) * D_;
  v8f ck = {}, cv = {};
#pragma unroll
  for (int k0 = 0; k0 < D_; k0 += 32) {
    v16h a = frag_ld(Arow, k0, half);
    ck = wmma16(a, frag_ld(Bk, k0, half), ck);
    cv = wmma16(a, frag_ld(Bv, k0, half), cv);
  }
  const float bkv = bk[colb + l16], bvv = bv[colb + l16];
  const int bt = row0 >> 12;                 // N_=4096
  const int n0 = row0 & (N_ - 1);
#pragma unroll
  for (int r = 0; r < 8; ++r) {
    int rr = r + half * 8;
    // k row-major: kh[(b,t,n), d]
    kh[(size_t)(row0 + rr) * D_ + colb + l16] = (_Float16)(ck[r] + bkv);
    // v transposed: vt[(b,t), d, n]
    vt[(size_t)bt * D_ * N_ + (size_t)(colb + l16) * N_ + n0 + rr] = (_Float16)(cv[r] + bvv);
  }
}

// ---- K3: q = LN(slots)@Wq^T + bq  ->  qh[bt][16][128] f16 (rows 8..15 zero) -
__global__ void k_qproj(const float* __restrict__ slots, const float* __restrict__ g,
                        const float* __restrict__ beta, const _Float16* __restrict__ WqH,
                        const float* __restrict__ bq, _Float16* __restrict__ qh) {
  __shared__ __align__(16) _Float16 lq[16 * PITCH_];
  const int tid = threadIdx.x, lane = tid & 31, wave = tid >> 5;
  const int l16 = lane & 15, half = lane >> 4;
  const int bt = blockIdx.x;
  _Float16* qbase = qh + (size_t)bt * 16 * D_;

  // zero pad rows 8..15 of qh for this (b,t)
  for (int i = tid; i < 8 * D_; i += 256) qbase[8 * D_ + i] = (_Float16)0.f;

  // LN of 8 slot rows: wave w -> row w, 32 lanes x 4 elems
  {
    const float* src = slots + (size_t)(bt * S_ + wave) * D_ + lane * 4;
    float v[4], s = 0.f, sq = 0.f;
#pragma unroll
    for (int c = 0; c < 4; ++c) { v[c] = src[c]; s += v[c]; sq += v[c] * v[c]; }
#pragma unroll
    for (int m = 1; m < 32; m <<= 1) { s += __shfl_xor(s, m, 32); sq += __shfl_xor(sq, m, 32); }
    float mean = s * (1.f / D_);
    float inv  = rsqrtf(sq * (1.f / D_) - mean * mean + LN_EPS_);
#pragma unroll
    for (int c = 0; c < 4; ++c) {
      int col = lane * 4 + c;
      lq[wave * PITCH_ + col] = (_Float16)((v[c] - mean) * inv * g[col] + beta[col]);
      lq[(wave + 8) * PITCH_ + col] = (_Float16)0.f;   // zero pad rows 8..15
    }
  }
  __syncthreads();

  const int colb = wave * 16;
  const _Float16* Arow = lq + l16 * PITCH_;
  const _Float16* Brow = WqH + (size_t)(colb + l16) * D_;
  v8f c = {};
#pragma unroll
  for (int k0 = 0; k0 < D_; k0 += 32) c = wmma16(frag_ld(Arow, k0, half), frag_ld(Brow, k0, half), c);
  const float bias = bq[colb + l16];
  if (half == 0) {
#pragma unroll
    for (int r = 0; r < 8; ++r) qbase[(size_t)r * D_ + colb + l16] = (_Float16)(c[r] + bias);
  }
}

// ---- K4: dots[bt][i][j] = scale * q[i,:] . k[j,:] --------------------------
__global__ void k_dots(const _Float16* __restrict__ qh, const _Float16* __restrict__ kh,
                       float* __restrict__ dots) {
  const int tid = threadIdx.x, lane = tid & 31, wave = tid >> 5;
  const int l16 = lane & 15, half = lane >> 4;
  const int task = blockIdx.x * 8 + wave;              // BT_ * (N_/16) tasks
  const int bt = task >> 8;                            // N_/16 = 256
  const int j0 = (task & 255) * 16;

  const _Float16* Arow = qh + (size_t)bt * 16 * D_ + (size_t)l16 * D_;
  const _Float16* Brow = kh + ((size_t)bt * N_ + j0 + l16) * D_;
  v8f c = {};
#pragma unroll
  for (int k0 = 0; k0 < D_; k0 += 32) c = wmma16(frag_ld(Arow, k0, half), frag_ld(Brow, k0, half), c);
  if (half == 0) {
#pragma unroll
    for (int r = 0; r < 8; ++r)
      dots[((size_t)bt * S_ + r) * N_ + j0 + l16] = c[r] * SCALE_;
  }
}

// ---- K5: softmax over t (axis=1) + EPS, in place ---------------------------
__global__ void k_softmax_t(float* __restrict__ dots) {
  int idx = blockIdx.x * 256 + threadIdx.x;            // over B_*S_*N_
  int j = idx & (N_ - 1);
  int i = (idx / N_) & (S_ - 1);
  int b = idx / (S_ * N_);
  float v[T_], m = -1e30f;
#pragma unroll
  for (int t = 0; t < T_; ++t) {
    v[t] = dots[(((size_t)b * T_ + t) * S_ + i) * N_ + j];
    m = fmaxf(m, v[t]);
  }
  float s = 0.f;
#pragma unroll
  for (int t = 0; t < T_; ++t) { v[t] = __expf(v[t] - m); s += v[t]; }
  float inv = 1.f / s;
#pragma unroll
  for (int t = 0; t < T_; ++t)
    dots[(((size_t)b * T_ + t) * S_ + i) * N_ + j] = v[t] * inv + EPS_;
}

// ---- K5c: deterministic per-row (over j) inverse sums ----------------------
__global__ void k_rowsum(const float* __restrict__ attn, float* __restrict__ rowinv) {
  __shared__ float part[8];
  const int row = blockIdx.x;                          // BT_*S_ rows
  const int tid = threadIdx.x, lane = tid & 31, wave = tid >> 5;
  const float* src = attn + (size_t)row * N_;
  float s = 0.f;
#pragma unroll
  for (int k = 0; k < N_ / 256; ++k) s += src[tid + k * 256];
#pragma unroll
  for (int m = 1; m < 32; m <<= 1) s += __shfl_xor(s, m, 32);
  if (lane == 0) part[wave] = s;
  __syncthreads();
  if (tid == 0) {
    float tot = 0.f;
#pragma unroll
    for (int w = 0; w < 8; ++w) tot += part[w];
    rowinv[row] = 1.f / tot;
  }
}

// ---- K5b: attn_f16 = attn * rowinv -----------------------------------------
__global__ void k_norm_cvt(const float* __restrict__ attn, const float* __restrict__ rowinv,
                           _Float16* __restrict__ attnh) {
  size_t idx = (size_t)blockIdx.x * 256 + threadIdx.x; // over BT_*S_*N_
  attnh[idx] = (_Float16)(attn[idx] * rowinv[idx / N_]);
}

// ---- K6a: split-K partials of updates = attn @ v ---------------------------
// task = ((chunk*BT_ + bt)*8 + dtile); each task: 16 WMMA steps over JCH_=512 j's
__global__ void k_updates_part(const _Float16* __restrict__ attnh, const _Float16* __restrict__ vt,
                               float* __restrict__ updp) {
  const int tid = threadIdx.x, lane = tid & 31, wave = tid >> 5;
  const int l16 = lane & 15, half = lane >> 4;
  const int task  = blockIdx.x * 8 + wave;             // NCH_*BT_*8 = 4096 tasks
  const int dtile = task & 7;
  const int bt    = (task >> 3) & (BT_ - 1);
  const int chunk = task >> 9;
  const int colb  = dtile * 16;
  const int jbase = chunk * JCH_;

  const bool arow_valid = (l16 < S_);
  const _Float16* Arow = attnh + ((size_t)bt * S_ + (arow_valid ? l16 : 0)) * N_ + jbase;
  const _Float16* Brow = vt + (size_t)bt * D_ * N_ + (size_t)(colb + l16) * N_ + jbase;
  const v16h zf = frag_zero();
  v8f c = {};
#pragma unroll 4
  for (int j0 = 0; j0 < JCH_; j0 += 32) {
    v16h a = arow_valid ? frag_ld(Arow, j0, half) : zf;
    c = wmma16(a, frag_ld(Brow, j0, half), c);
  }
  if (half == 0) {
#pragma unroll
    for (int r = 0; r < 8; ++r)
      updp[(((size_t)chunk * BT_ + bt) * S_ + r) * D_ + colb + l16] = c[r];
  }
}

// ---- K6b: reduce the NCH_ partials ------------------------------------------
__global__ void k_upd_reduce(const float* __restrict__ updp, float* __restrict__ upd) {
  int idx = blockIdx.x * 256 + threadIdx.x;            // over BT_*S_*D_ = 65536
  float s = 0.f;
#pragma unroll
  for (int c = 0; c < NCH_; ++c) s += updp[(size_t)c * BT_ * S_ * D_ + idx];
  upd[idx] = s;
}

// ---- K7: GRU cell + LN + MLP residual, one block per (b,t) ------------------
__global__ void k_gru_mlp(const float* __restrict__ upd, float* __restrict__ slots,
                          const _Float16* __restrict__ WihH, const _Float16* __restrict__ WhhH,
                          const float* __restrict__ b_ih, const float* __restrict__ b_hh,
                          const _Float16* __restrict__ W1H, const float* __restrict__ b1,
                          const _Float16* __restrict__ W2H, const float* __restrict__ b2,
                          const float* __restrict__ g_ff, const float* __restrict__ beta_ff,
                          float* __restrict__ out, int write_out) {
  __shared__ __align__(16) _Float16 uh[16 * PITCH_];
  __shared__ __align__(16) _Float16 sh[16 * PITCH_];
  __shared__ __align__(16) _Float16 lnh[16 * PITCH_];
  __shared__ __align__(16) _Float16 h1h[16 * PITCH_];
  __shared__ float gi[S_ * 3 * D_];
  __shared__ float gh[S_ * 3 * D_];
  __shared__ float ns[S_ * D_];

  const int tid = threadIdx.x, lane = tid & 31, wave = tid >> 5;
  const int l16 = lane & 15, half = lane >> 4;
  const int bt = blockIdx.x;
  const float* updB = upd + (size_t)bt * S_ * D_;
  const float* slB  = slots + (size_t)bt * S_ * D_;

  // Phase A: stage f16 operands, zero pad rows 8..15 everywhere
  for (int e = tid; e < S_ * D_; e += 256) {
    int i = e >> 7, dd = e & 127;
    uh[i * PITCH_ + dd] = (_Float16)updB[e];
    sh[i * PITCH_ + dd] = (_Float16)slB[e];
    uh[(i + 8) * PITCH_ + dd] = (_Float16)0.f;
    sh[(i + 8) * PITCH_ + dd] = (_Float16)0.f;
    lnh[(i + 8) * PITCH_ + dd] = (_Float16)0.f;
    h1h[(i + 8) * PITCH_ + dd] = (_Float16)0.f;
  }
  __syncthreads();

  // Phase B: gi = upd@Wih^T + b_ih ; gh = h@Whh^T + b_hh   (48 16-col tiles)
  for (int task = wave; task < 48; task += 8) {
    const int which = task / 24;                       // 0: gi, 1: gh
    const int ct = task % 24;
    const _Float16* Arow = (which ? sh : uh) + l16 * PITCH_;
    const _Float16* Brow = (which ? WhhH : WihH) + (size_t)(ct * 16 + l16) * D_;
    v8f c = {};
#pragma unroll
    for (int k0 = 0; k0 < D_; k0 += 32) c = wmma16(frag_ld(Arow, k0, half), frag_ld(Brow, k0, half), c);
    const float bias = (which ? b_hh : b_ih)[ct * 16 + l16];
    if (half == 0) {
      float* dst = which ? gh : gi;
#pragma unroll
      for (int r = 0; r < 8; ++r) dst[r * (3 * D_) + ct * 16 + l16] = c[r] + bias;
    }
  }
  __syncthreads();

  // Phase C: GRU gates (f32, exact formula)
  for (int e = tid; e < S_ * D_; e += 256) {
    int i = e >> 7, dd = e & 127;
    float i_r = gi[i * 384 + dd],       h_r = gh[i * 384 + dd];
    float i_z = gi[i * 384 + 128 + dd], h_z = gh[i * 384 + 128 + dd];
    float i_n = gi[i * 384 + 256 + dd], h_n = gh[i * 384 + 256 + dd];
    float r = sigmoidf_(i_r + h_r);
    float z = sigmoidf_(i_z + h_z);
    float n = tanhf(i_n + r * h_n);
    ns[e] = (1.f - z) * n + z * slB[e];
  }
  __syncthreads();

  // Phase D: LN(ns) with g_ff/beta_ff -> lnh (f16), wave w -> row w
  {
    float v[4], s = 0.f, sq = 0.f;
#pragma unroll
    for (int c = 0; c < 4; ++c) { v[c] = ns[wave * D_ + lane * 4 + c]; s += v[c]; sq += v[c] * v[c]; }
#pragma unroll
    for (int m = 1; m < 32; m <<= 1) { s += __shfl_xor(s, m, 32); sq += __shfl_xor(sq, m, 32); }
    float mean = s * (1.f / D_);
    float inv  = rsqrtf(sq * (1.f / D_) - mean * mean + LN_EPS_);
#pragma unroll
    for (int c = 0; c < 4; ++c) {
      int col = lane * 4 + c;
      lnh[wave * PITCH_ + col] = (_Float16)((v[c] - mean) * inv * g_ff[col] + beta_ff[col]);
    }
  }
  __syncthreads();

  // Phase E: h1 = relu(lnh @ W1^T + b1) -> h1h
  {
    const int colb = wave * 16;
    const _Float16* Arow = lnh + l16 * PITCH_;
    const _Float16* Brow = W1H + (size_t)(colb + l16) * D_;
    v8f c = {};
#pragma unroll
    for (int k0 = 0; k0 < D_; k0 += 32) c = wmma16(frag_ld(Arow, k0, half), frag_ld(Brow, k0, half), c);
    const float bias = b1[colb + l16];
    if (half == 0) {
#pragma unroll
      for (int r = 0; r < 8; ++r) h1h[r * PITCH_ + colb + l16] = (_Float16)fmaxf(0.f, c[r] + bias);
    }
  }
  __syncthreads();

  // Phase F: slots = ns + h1 @ W2^T + b2
  {
    const int colb = wave * 16;
    const _Float16* Arow = h1h + l16 * PITCH_;
    const _Float16* Brow = W2H + (size_t)(colb + l16) * H_;
    v8f c = {};
#pragma unroll
    for (int k0 = 0; k0 < H_; k0 += 32) c = wmma16(frag_ld(Arow, k0, half), frag_ld(Brow, k0, half), c);
    const float bias = b2[colb + l16];
    if (half == 0) {
#pragma unroll
      for (int r = 0; r < 8; ++r) {
        float val = ns[r * D_ + colb + l16] + c[r] + bias;
        size_t o = ((size_t)bt * S_ + r) * D_ + colb + l16;
        slots[o] = val;
        if (write_out) out[o] = val;
      }
    }
  }
}

// ----------------------------------------------------------------------------
extern "C" void kernel_launch(void* const* d_in, const int* in_sizes, int n_in,
                              void* d_out, int out_size, void* d_ws, size_t ws_size,
                              hipStream_t stream) {
  const float* inputs  = (const float*)d_in[0];
  const float* noise   = (const float*)d_in[1];
  const float* mu      = (const float*)d_in[2];
  const float* lsig    = (const float*)d_in[3];
  const float* Wq = (const float*)d_in[4];  const float* bq = (const float*)d_in[5];
  const float* Wk = (const float*)d_in[6];  const float* bk = (const float*)d_in[7];
  const float* Wv = (const float*)d_in[8];  const float* bv = (const float*)d_in[9];
  const float* W_ih = (const float*)d_in[10]; const float* W_hh = (const float*)d_in[11];
  const float* b_ih = (const float*)d_in[12]; const float* b_hh = (const float*)d_in[13];
  const float* W1 = (const float*)d_in[14]; const float* b1 = (const float*)d_in[15];
  const float* W2 = (const float*)d_in[16]; const float* b2 = (const float*)d_in[17];
  const float* g_in = (const float*)d_in[18]; const float* beta_in = (const float*)d_in[19];
  const float* g_sl = (const float*)d_in[20]; const float* beta_sl = (const float*)d_in[21];
  const float* g_ff = (const float*)d_in[22]; const float* beta_ff = (const float*)d_in[23];
  float* out = (float*)d_out;

  char* ws = (char*)d_ws;
  size_t off = 0;
  auto take = [&](size_t bytes) -> char* {
    char* p = ws + off;
    off = (off + bytes + 255) & ~(size_t)255;
    return p;
  };
  _Float16* kh    = (_Float16*)take(sizeof(_Float16) * (size_t)BT_ * N_ * D_);  // 64 MB
  _Float16* vt    = (_Float16*)take(sizeof(_Float16) * (size_t)BT_ * D_ * N_);  // 64 MB (transposed)
  float*    dots  = (float*)   take(sizeof(float)    * (size_t)BT_ * S_ * N_);  // 8 MB
  _Float16* attnh = (_Float16*)take(sizeof(_Float16) * (size_t)BT_ * S_ * N_);  // 4 MB
  _Float16* qh    = (_Float16*)take(sizeof(_Float16) * (size_t)BT_ * 16 * D_);
  float*    slotsG= (float*)   take(sizeof(float)    * (size_t)BT_ * S_ * D_);
  float*    updG  = (float*)   take(sizeof(float)    * (size_t)BT_ * S_ * D_);
  float*    updP  = (float*)   take(sizeof(float)    * (size_t)NCH_ * BT_ * S_ * D_); // 2 MB
  float*    rowinv= (float*)   take(sizeof(float)    * (size_t)BT_ * S_);
  _Float16* WqH  = (_Float16*)take(sizeof(_Float16) * D_ * D_);
  _Float16* WkH  = (_Float16*)take(sizeof(_Float16) * D_ * D_);
  _Float16* WvH  = (_Float16*)take(sizeof(_Float16) * D_ * D_);
  _Float16* WihH = (_Float16*)take(sizeof(_Float16) * 3 * D_ * D_);
  _Float16* WhhH = (_Float16*)take(sizeof(_Float16) * 3 * D_ * D_);
  _Float16* W1H  = (_Float16*)take(sizeof(_Float16) * H_ * D_);
  _Float16* W2H  = (_Float16*)take(sizeof(_Float16) * D_ * H_);

  // K0: weights to f16
  k_cvt_f16<<<(D_ * D_ + 255) / 256, 256, 0, stream>>>(Wq, WqH, D_ * D_);
  k_cvt_f16<<<(D_ * D_ + 255) / 256, 256, 0, stream>>>(Wk, WkH, D_ * D_);
  k_cvt_f16<<<(D_ * D_ + 255) / 256, 256, 0, stream>>>(Wv, WvH, D_ * D_);
  k_cvt_f16<<<(3 * D_ * D_ + 255) / 256, 256, 0, stream>>>(W_ih, WihH, 3 * D_ * D_);
  k_cvt_f16<<<(3 * D_ * D_ + 255) / 256, 256, 0, stream>>>(W_hh, WhhH, 3 * D_ * D_);
  k_cvt_f16<<<(H_ * D_ + 255) / 256, 256, 0, stream>>>(W1, W1H, H_ * D_);
  k_cvt_f16<<<(D_ * H_ + 255) / 256, 256, 0, stream>>>(W2, W2H, D_ * H_);

  // K1: slot init
  k_init_slots<<<(BT_ * S_ * D_) / 256, 256, 0, stream>>>(noise, mu, lsig, slotsG);

  // K2: fused LN + k/v projection (the one HBM-heavy pass)
  k_ln_kv<<<BT_ * N_ / 16, 256, 0, stream>>>(inputs, g_in, beta_in, WkH, WvH, bk, bv, kh, vt);

  for (int it = 0; it < ITERS_; ++it) {
    k_qproj<<<BT_, 256, 0, stream>>>(slotsG, g_sl, beta_sl, WqH, bq, qh);
    k_dots<<<BT_ * (N_ / 16) / 8, 256, 0, stream>>>(qh, kh, dots);
    k_softmax_t<<<(B_ * S_ * N_) / 256, 256, 0, stream>>>(dots);
    k_rowsum<<<BT_ * S_, 256, 0, stream>>>(dots, rowinv);
    k_norm_cvt<<<(BT_ * S_ * N_) / 256, 256, 0, stream>>>(dots, rowinv, attnh);
    k_updates_part<<<(NCH_ * BT_ * 8) / 8, 256, 0, stream>>>(attnh, vt, updP);
    k_upd_reduce<<<(BT_ * S_ * D_) / 256, 256, 0, stream>>>(updP, updG);
    k_gru_mlp<<<BT_, 256, 0, stream>>>(updG, slotsG, WihH, WhhH, b_ih, b_hh,
                                       W1H, b1, W2H, b2, g_ff, beta_ff,
                                       out, (it == ITERS_ - 1) ? 1 : 0);
  }
}
